// CustomExpert_2121713845122
// MI455X (gfx1250) — matrix operand order repaired and checked by reading, compiled, not run on hardware
//
#include <hip/hip_runtime.h>
#include <hip/hip_bf16.h>

// Problem constants (match reference): E=4, T=4096, D=2048, H=8192.
constexpr int EE = 4, TT = 4096, DD = 2048, HH = 8192;

#define BM  128
#define BN  128
#define BK  64
#define LDK 72   // BK + 8 pad ushorts -> 144B row stride: bank-conflict free, 16B aligned

typedef __attribute__((ext_vector_type(16))) __bf16 v16bf;
typedef __attribute__((ext_vector_type(8)))  float  v8f;

union Frag {
    v16bf v;
    uint4 q[2];
};

// Packed f32->bf16 (RNE) via the hardware cvt op: one VOP3 per two elements.
// (__builtin_amdgcn_cvt_pk_bf16_f32 is not declared by this toolchain, so emit it directly.)
__device__ __forceinline__ unsigned int f2bf_pk(float a, float b) {
    unsigned int r;
    asm("v_cvt_pk_bf16_f32 %0, %1, %2" : "=v"(r) : "v"(a), "v"(b));
    return r;
}

// CDNA5 async memory->LDS copy (no VGPR round trip, tracked by ASYNCcnt).
// HW adds the wave's LDS_BASE to the byte offset we pass in VDST.
__device__ __forceinline__ void async_ld_b128(unsigned int lds_byte_off, const void* gaddr) {
    asm volatile("global_load_async_to_lds_b128 %0, %1, off"
                 :: "v"(lds_byte_off), "v"(gaddr) : "memory");
}
__device__ __forceinline__ void wait_async0() {
    asm volatile("s_wait_asynccnt 0x0" ::: "memory");
}

// C[M,N] = act(A[M,K] @ B[K,N] + bias), per expert (blockIdx.z).
// A_BF16: A is bf16 (ws) and is moved with async-to-LDS.  RELU_BF16_OUT: relu + bf16 store.
template<bool A_BF16, bool RELU_BF16_OUT>
__global__ __launch_bounds__(256)
void ffn_gemm_kernel(const void* __restrict__ Aptr,
                     const float* __restrict__ Bptr,
                     const float* __restrict__ biasPtr,
                     void* __restrict__ Cptr,
                     int M, int N, int K)
{
    // Single LDS allocation so LDS byte offset == 2 * element offset (for async copies).
    __shared__ unsigned short smem[(2 * BM + 2 * BN) * LDK];

    auto A_idx = [](int buf, int row, int k) { return (buf * BM + row) * LDK + k; };
    auto B_idx = [](int buf, int row, int k) { return (2 * BM + buf * BN + row) * LDK + k; };

    const int e    = blockIdx.z;
    const int mblk = blockIdx.y * BM;
    const int nblk = blockIdx.x * BN;
    const int tid  = threadIdx.x;
    const int wave = tid >> 5;
    const int lane = tid & 31;
    const int half = lane >> 4;     // which 16-lane half of the wave
    const int r    = lane & 15;
    const int wmb  = (wave & 1) * 64;   // wave M offset inside tile (2 waves in M)
    const int wnb  = (wave >> 1) * 32;  // wave N offset inside tile (4 waves in N)

    const float* Bg = Bptr + (size_t)e * K * N + nblk;

    float4 aRegF[8];   // f32 A prefetch (pass 1 only)
    float4 bReg[8];

    // ---- B: global f32 -> registers for K-tile `kt` ----
    auto gloadB = [&](int kt) {
        const int kb = kt * BK;
        const int nv = (tid & 31) * 4;
        const int k0 = tid >> 5;
        #pragma unroll
        for (int p = 0; p < 8; ++p)
            bReg[p] = *(const float4*)(Bg + (size_t)(kb + k0 + p * 8) * N + nv);
    };
    // ---- B: registers -> LDS bf16, transposed to [n][k] ----
    auto ldsStoreB = [&](int buf) {
        const int nv = (tid & 31) * 4;
        const int k0 = tid >> 5;
        #pragma unroll
        for (int p = 0; p < 8; ++p) {
            const int k = k0 + p * 8;
            float4 v = bReg[p];
            unsigned int lo = f2bf_pk(v.x, v.y);
            unsigned int hi = f2bf_pk(v.z, v.w);
            smem[B_idx(buf, nv + 0, k)] = (unsigned short)lo;
            smem[B_idx(buf, nv + 1, k)] = (unsigned short)(lo >> 16);
            smem[B_idx(buf, nv + 2, k)] = (unsigned short)hi;
            smem[B_idx(buf, nv + 3, k)] = (unsigned short)(hi >> 16);
        }
    };

    // ---- A (pass 1, f32): global -> regs -> convert -> LDS ----
    auto gloadA = [&](int kt) {
        const int kb  = kt * BK;
        const float* Ag = (const float*)Aptr + ((size_t)e * M + mblk) * K + kb;
        const int kv  = (tid & 15) * 4;
        const int row = tid >> 4;
        #pragma unroll
        for (int p = 0; p < 8; ++p)
            aRegF[p] = *(const float4*)(Ag + (size_t)(row + p * 16) * K + kv);
    };
    auto ldsStoreA = [&](int buf) {
        const int kv  = (tid & 15) * 4;
        const int row = tid >> 4;
        #pragma unroll
        for (int p = 0; p < 8; ++p) {
            float4 v = aRegF[p];
            uint2 pk;
            pk.x = f2bf_pk(v.x, v.y);
            pk.y = f2bf_pk(v.z, v.w);
            *(uint2*)&smem[A_idx(buf, row + p * 16, kv)] = pk;
        }
    };

    // ---- A (pass 2, bf16): async global -> LDS, no VGPR round trip ----
    auto issueAsyncA = [&](int kt) {
        const int kb  = kt * BK;
        const unsigned short* Ag =
            (const unsigned short*)Aptr + ((size_t)e * M + mblk) * K + kb;
        const int kv  = (tid & 7) * 8;
        const int row = tid >> 3;
        const int buf = kt & 1;
        #pragma unroll
        for (int p = 0; p < 4; ++p) {
            unsigned int loff = 2u * (unsigned int)A_idx(buf, row + p * 32, kv);
            async_ld_b128(loff, Ag + (size_t)(row + p * 32) * K + kv);
        }
    };

    v8f acc[4][2] = {};

    // ---- one BK=64 tile: two k=32 WMMA steps ----
    auto compute = [&](int buf) {
        #pragma unroll
        for (int ks = 0; ks < 2; ++ks) {
            const int kb = ks * 32;
            Frag a[4], b[2];
            #pragma unroll
            for (int mi = 0; mi < 4; ++mi) {
                // A 16x32 bf16: lane half 0 holds K 0..15, half 1 holds K 16..31
                const unsigned short* s = &smem[A_idx(buf, wmb + mi * 16 + r, kb + half * 16)];
                a[mi].q[0] = *(const uint4*)(s);        // K +0..7
                a[mi].q[1] = *(const uint4*)(s + 8);    // K +8..15
            }
            #pragma unroll
            for (int ni = 0; ni < 2; ++ni) {
                // B 32x16 bf16: lane column N=r; half 0 -> K 0..15, half 1 -> K 16..31
                const unsigned short* s = &smem[B_idx(buf, wnb + ni * 16 + r, kb + half * 16)];
                b[ni].q[0] = *(const uint4*)(s);        // K +0..7
                b[ni].q[1] = *(const uint4*)(s + 8);    // K +8..15
            }
            #pragma unroll
            for (int mi = 0; mi < 4; ++mi)
                #pragma unroll
                for (int ni = 0; ni < 2; ++ni)
                    acc[mi][ni] = __builtin_amdgcn_wmma_f32_16x16x32_bf16(
                        false, a[mi].v, false, b[ni].v,
                        (short)0, acc[mi][ni], false, false);
        }
    };

    const int NT = K / BK;
    if constexpr (A_BF16) {
        gloadB(0);
        issueAsyncA(0);
        for (int t = 0; t < NT; ++t) {
            ldsStoreB(t & 1);
            wait_async0();            // this wave's tile-t async copies landed
            __syncthreads();          // all waves' tile-t data visible
            if (t + 1 < NT) {
                gloadB(t + 1);
                issueAsyncA(t + 1);   // overlaps with WMMA below (other buffer)
            }
            compute(t & 1);
        }
    } else {
        gloadA(0);
        gloadB(0);
        for (int t = 0; t < NT; ++t) {
            ldsStoreA(t & 1);
            ldsStoreB(t & 1);
            __syncthreads();
            if (t + 1 < NT) {
                gloadA(t + 1);
                gloadB(t + 1);
            }
            compute(t & 1);
        }
    }

    // ---- epilogue: bias (+relu) and store ----
    #pragma unroll
    for (int ni = 0; ni < 2; ++ni) {
        const int n = nblk + wnb + ni * 16 + r;
        const float bv = biasPtr[(size_t)e * N + n];
        #pragma unroll
        for (int mi = 0; mi < 4; ++mi) {
            const int mb = mblk + wmb + mi * 16 + half * 8;
            if constexpr (RELU_BF16_OUT) {
                #pragma unroll
                for (int i = 0; i < 8; i += 2) {
                    float v0 = fmaxf(acc[mi][ni][i]     + bv, 0.0f);
                    float v1 = fmaxf(acc[mi][ni][i + 1] + bv, 0.0f);
                    unsigned int pk = f2bf_pk(v0, v1);
                    unsigned short* out = (unsigned short*)Cptr;
                    out[((size_t)e * M + mb + i)     * N + n] = (unsigned short)pk;
                    out[((size_t)e * M + mb + i + 1) * N + n] = (unsigned short)(pk >> 16);
                }
            } else {
                #pragma unroll
                for (int i = 0; i < 8; ++i) {
                    ((float*)Cptr)[((size_t)e * M + mb + i) * N + n] = acc[mi][ni][i] + bv;
                }
            }
        }
    }
}

extern "C" void kernel_launch(void* const* d_in, const int* in_sizes, int n_in,
                              void* d_out, int out_size, void* d_ws, size_t ws_size,
                              hipStream_t stream) {
    const float* x   = (const float*)d_in[0];   // [E,T,D]
    const float* w1  = (const float*)d_in[1];   // [E,D,H]
    const float* b1  = (const float*)d_in[2];   // [E,1,H]
    const float* w2  = (const float*)d_in[3];   // [E,H,D]
    const float* b2  = (const float*)d_in[4];   // [E,1,D]
    (void)in_sizes; (void)n_in; (void)out_size; (void)ws_size;

    // Intermediate relu(x@W1+b1) kept as bf16 in workspace: E*T*H*2 = 256 MB.
    unsigned short* ws = (unsigned short*)d_ws;

    dim3 blk(256);
    // Pass 1: [T,D]@[D,H] -> relu -> bf16 ws
    dim3 g1(HH / BN, TT / BM, EE);
    ffn_gemm_kernel<false, true><<<g1, blk, 0, stream>>>(
        x, w1, b1, ws, TT, HH, DD);
    // Pass 2: [T,H]@[H,D] -> f32 out
    dim3 g2(DD / BN, TT / BM, EE);
    ffn_gemm_kernel<true, false><<<g2, blk, 0, stream>>>(
        ws, w2, b2, d_out, TT, DD, HH);
}
